// Aligner_38783554682996
// MI455X (gfx1250) — matrix operand order repaired and checked by reading, not compiled, not run on hardware
//
#include <hip/hip_runtime.h>

typedef __bf16 bf16_t;
typedef __attribute__((ext_vector_type(16))) __bf16 v16bf;
typedef __attribute__((ext_vector_type(8)))  __bf16 v8bf;
typedef __attribute__((ext_vector_type(8)))  float  v8f;

static constexpr int NWG  = 64;    // persistent workgroups (== batch)
static constexpr int NTHR = 256;   // 8 wave32 per WG
static constexpr int B_ = 64, S_ = 512, I_ = 512, T_ = 1024, H_ = 80, C_ = 512;
static constexpr int KA_  = 608;   // padded K for w_ih GEMM (80 frame + 512 prev + 16 pad)
static constexpr int LDX_ = 1120;  // x row stride: 608 + 512 (h)
static constexpr int N3C_ = 1536;  // 3*C

__device__ __forceinline__ v16bf cat8(v8bf lo, v8bf hi){
  v16bf r;
#pragma unroll
  for (int i = 0; i < 8; ++i){ r[i] = lo[i]; r[i+8] = hi[i]; }
  return r;
}

// A fragment (16x32 bf16, M x K): lane&15 = M row; lanes>=16 hold K+8 .. / K+24 ..
// VGPR0-3 = K kb..kb+7, VGPR4-7 = K kb+16..kb+23, kb = (lane>=16)?8:0.
__device__ __forceinline__ v16bf load_a_frag(const bf16_t* p, int lane){
  const bf16_t* q = p + ((lane >> 4) << 3);
  v8bf lo = *(const v8bf*)(q);        // two global_load_b128 from L2
  v8bf hi = *(const v8bf*)(q + 16);
  return cat8(lo, hi);
}

// B fragment (32x16 bf16, K x N): lane&15 = N col; VGPR0-3 = K kb..kb+7, VGPR4-7 = K kb+16..kb+23, kb = (lane>=16)?8:0.
__device__ __forceinline__ v16bf load_b_frag(const bf16_t* p, int lane){
  const bf16_t* q = p + ((lane >> 4) << 3);
  v8bf lo = *(const v8bf*)(q);
  v8bf hi = *(const v8bf*)(q + 16);
  return cat8(lo, hi);
}

__device__ __forceinline__ float sigmoidf_(float x){ return 1.f / (1.f + expf(-x)); }

// Device-wide barrier for the persistent grid (all NWG WGs co-resident).
__device__ void grid_sync(unsigned* cnt, unsigned* gen){
  __syncthreads();
  if (threadIdx.x == 0){
    __threadfence();
    unsigned g = atomicAdd(gen, 0u);
    unsigned a = atomicAdd(cnt, 1u);
    if (a == (unsigned)(NWG - 1)){
      atomicExch(cnt, 0u);
      __threadfence();
      atomicAdd(gen, 1u);
    } else {
      while (atomicAdd(gen, 0u) == g) __builtin_amdgcn_s_sleep(2);
    }
    __threadfence();
  }
  __syncthreads();
}

// -------- prep kernels (run once per call; everything becomes L2-resident bf16) --------
__global__ void k_cast_bf16(const float* __restrict__ src, bf16_t* __restrict__ dst, int n){
  int i = blockIdx.x * 256 + threadIdx.x;
  if (i < n) dst[i] = (bf16_t)src[i];
}
__global__ void k_prep_wih(const float* __restrict__ w_ih, bf16_t* __restrict__ WgA){
  int i = blockIdx.x * 256 + threadIdx.x;
  if (i < N3C_ * KA_){
    int n = i / KA_, k = i % KA_;
    WgA[i] = (bf16_t)((k < H_ + I_) ? w_ih[n * (H_ + I_) + k] : 0.f);
  }
}
__global__ void k_init(bf16_t* __restrict__ xbf, float* __restrict__ Ht, unsigned* __restrict__ bar){
  int i = blockIdx.x * 256 + threadIdx.x;
  if (i < B_ * LDX_) xbf[i] = (bf16_t)0.f;   // h0 = 0, pad region = 0
  if (i < C_ * B_)   Ht[i]  = 0.f;
  if (i < 2)         bar[i] = 0u;
}

// -------- persistent aligner: 1024 sequential GRU/attention steps --------
__global__ __launch_bounds__(NTHR) void aligner_persistent(
    const float* __restrict__ gt,  const float* __restrict__ mask,
    const float* __restrict__ b_ih, const float* __restrict__ b_hh,
    const float* __restrict__ b1,  const float* __restrict__ w2, const float* __restrict__ b2,
    const bf16_t* __restrict__ encbf, const bf16_t* __restrict__ WgA,
    const bf16_t* __restrict__ WgB,   const bf16_t* __restrict__ w1bf,
    bf16_t* __restrict__ xbf, float* __restrict__ G,
    float* __restrict__ Ht,   float* __restrict__ Yt,
    unsigned* __restrict__ bar, float* __restrict__ out)
{
  __shared__ float s_alpha[S_];       // alpha row for the batch this WG owns
  __shared__ float s_red[2 * NTHR];

  const int tid  = threadIdx.x;
  const int g    = blockIdx.x;
  const int lane = tid & 31;
  const int waveFlat = g * (NTHR / 32) + (tid >> 5);
  const int b = g;                    // batch owned by this WG (phases A and D)

  for (int s = tid; s < S_; s += NTHR) s_alpha[s] = (s == 0) ? 1.f : 0.f;
  __syncthreads();

  for (int t = 0; t < T_; ++t){
    // ---- Phase A: x = [frame | prev | 0 | h];  prev = alpha @ enc[b]  (VALU, L2 bound) ----
    for (int k = tid; k < H_; k += NTHR)
      xbf[(size_t)b * LDX_ + k] = (bf16_t)gt[((size_t)b * T_ + t) * H_ + k];
    {
      float acc0 = 0.f, acc1 = 0.f;
      const bf16_t* eb = encbf + (size_t)b * S_ * I_;
      const int i0 = tid, i1 = tid + NTHR;
      for (int s = 0; s < S_; ++s){
        float a = s_alpha[s];
        if (a != 0.f){                       // alpha is sparse early on
          acc0 += a * (float)eb[s * I_ + i0];
          acc1 += a * (float)eb[s * I_ + i1];
        }
      }
      xbf[(size_t)b * LDX_ + H_ + i0] = (bf16_t)acc0;
      xbf[(size_t)b * LDX_ + H_ + i1] = (bf16_t)acc1;
    }
    grid_sync(bar, bar + 1);

    // ---- Phase B (WMMA): G[0:1536][b]   = w_ih · xᵀ  (K=608)
    //                      G[1536:3072][b]= w_hh · hᵀ  (K=512)   f32 accum ----
    for (int st = waveFlat; st < 768; st += NWG * (NTHR / 32)){
      const bool sel = (st >= 384);
      const int stt = sel ? st - 384 : st;
      const int nt = stt >> 2, bt = stt & 3;
      const int ldA = sel ? C_ : KA_;
      const bf16_t* Arow = (sel ? WgB : WgA) + (size_t)(nt * 16 + (lane & 15)) * ldA;
      const bf16_t* Xrow = xbf + (size_t)(bt * 16 + (lane & 15)) * LDX_ + (sel ? KA_ : 0);
      const int nkt = sel ? (C_ / 32) : (KA_ / 32);
      v8f acc = {0.f,0.f,0.f,0.f,0.f,0.f,0.f,0.f};
      for (int kt = 0; kt < nkt; ++kt){
        v16bf af  = load_a_frag(Arow + kt * 32, lane);
        v16bf bfr = load_b_frag(Xrow + kt * 32, lane);
        acc = __builtin_amdgcn_wmma_f32_16x16x32_bf16(false, af, false, bfr,
                                                      (short)0, acc, false, false);
      }
      const int nn = lane & 15, mo = (lane >> 4) * 8;
      const int nbase = (sel ? N3C_ : 0) + nt * 16;
#pragma unroll
      for (int r = 0; r < 8; ++r)
        G[(size_t)(nbase + mo + r) * B_ + bt * 16 + nn] = acc[r];
    }
    grid_sync(bar, bar + 1);

    // ---- GRU gates (elementwise, coalesced over batch) ----
    for (int idx = g * NTHR + tid; idx < C_ * B_; idx += NWG * NTHR){
      const int c = idx >> 6, bb = idx & 63;
      float ir  = G[(size_t)c * B_ + bb]                 + b_ih[c];
      float iz  = G[(size_t)(C_ + c) * B_ + bb]          + b_ih[C_ + c];
      float in_ = G[(size_t)(2 * C_ + c) * B_ + bb]      + b_ih[2 * C_ + c];
      float hr  = G[(size_t)(N3C_ + c) * B_ + bb]        + b_hh[c];
      float hz  = G[(size_t)(N3C_ + C_ + c) * B_ + bb]   + b_hh[C_ + c];
      float hn  = G[(size_t)(N3C_ + 2 * C_ + c) * B_ + bb] + b_hh[2 * C_ + c];
      float r = sigmoidf_(ir + hr);
      float z = sigmoidf_(iz + hz);
      float n = tanhf(in_ + r * hn);
      float h = Ht[(size_t)c * B_ + bb];
      float hnew = (1.f - z) * n + z * h;
      Ht[(size_t)c * B_ + bb] = hnew;
      xbf[(size_t)bb * LDX_ + KA_ + c] = (bf16_t)hnew;   // feeds GEMMs of phase C / next step
    }
    grid_sync(bar, bar + 1);

    // ---- Phase C (WMMA): Yt = tanh(w1 · h_newᵀ + b1) ----
    if (waveFlat < 128){
      const int nt = waveFlat >> 2, bt = waveFlat & 3;
      const bf16_t* Arow = w1bf + (size_t)(nt * 16 + (lane & 15)) * C_;
      const bf16_t* Xrow = xbf + (size_t)(bt * 16 + (lane & 15)) * LDX_ + KA_;
      v8f acc = {0.f,0.f,0.f,0.f,0.f,0.f,0.f,0.f};
      for (int kt = 0; kt < C_ / 32; ++kt){
        v16bf af  = load_a_frag(Arow + kt * 32, lane);
        v16bf bfr = load_b_frag(Xrow + kt * 32, lane);
        acc = __builtin_amdgcn_wmma_f32_16x16x32_bf16(false, af, false, bfr,
                                                      (short)0, acc, false, false);
      }
      const int nn = lane & 15, mo = (lane >> 4) * 8;
#pragma unroll
      for (int r = 0; r < 8; ++r){
        int n = nt * 16 + mo + r;
        Yt[(size_t)n * B_ + bt * 16 + nn] = tanhf(acc[r] + b1[n]);
      }
    }
    grid_sync(bar, bar + 1);

    // ---- Phase D: 2-logit head, softmax, alpha recurrence (WG-private in LDS) ----
    {
      float l0 = 0.f, l1 = 0.f;
      for (int c = tid; c < C_; c += NTHR){
        float y = Yt[(size_t)c * B_ + b];
        l0 += y * w2[c];
        l1 += y * w2[C_ + c];
      }
      s_red[tid] = l0; s_red[NTHR + tid] = l1;
      __syncthreads();
      for (int off = NTHR / 2; off > 0; off >>= 1){
        if (tid < off){
          s_red[tid]        += s_red[tid + off];
          s_red[NTHR + tid] += s_red[NTHR + tid + off];
        }
        __syncthreads();
      }
      float L0 = s_red[0] + b2[0], L1 = s_red[NTHR] + b2[1];
      float m = fmaxf(L0, L1);
      float e0 = expf(L0 - m), e1 = expf(L1 - m);
      float inv = 1.f / (e0 + e1);
      float stop = e0 * inv, nxt = e1 * inv;

      float a0  = s_alpha[tid];
      float a0m = (tid > 0) ? s_alpha[tid - 1] : 0.f;
      float a1  = s_alpha[tid + NTHR];
      float a1m = s_alpha[tid + NTHR - 1];
      __syncthreads();
      float v0 = (stop * a0 + nxt * a0m) * mask[b * S_ + tid];
      float v1 = (stop * a1 + nxt * a1m) * mask[b * S_ + tid + NTHR];
      s_alpha[tid]        = v0;
      s_alpha[tid + NTHR] = v1;
      out[((size_t)b * T_ + t) * S_ + tid]        = v0;
      out[((size_t)b * T_ + t) * S_ + tid + NTHR] = v1;
      __syncthreads();
    }
    // No grid barrier here: alpha is WG-private; xbf[0:592) rewritten only by the
    // owning WG in next Phase A, and consumers wait at the next grid_sync.
  }
}

extern "C" void kernel_launch(void* const* d_in, const int* in_sizes, int n_in,
                              void* d_out, int out_size, void* d_ws, size_t ws_size,
                              hipStream_t stream){
  const float* enc  = (const float*)d_in[0];
  const float* mask = (const float*)d_in[1];
  const float* gt   = (const float*)d_in[2];
  const float* w_ih = (const float*)d_in[3];
  const float* w_hh = (const float*)d_in[4];
  const float* b_ih = (const float*)d_in[5];
  const float* b_hh = (const float*)d_in[6];
  const float* w1   = (const float*)d_in[7];
  const float* b1   = (const float*)d_in[8];
  const float* w2   = (const float*)d_in[9];
  const float* b2   = (const float*)d_in[10];
  float* out = (float*)d_out;
  (void)in_sizes; (void)n_in; (void)out_size; (void)ws_size;

  char* ws = (char*)d_ws;
  size_t off = 0;
  auto take = [&](size_t bytes)->char*{
    char* p = ws + off;
    off = (off + bytes + 255) & ~(size_t)255;
    return p;
  };
  bf16_t* encbf = (bf16_t*)take((size_t)B_ * S_ * I_ * 2);   // 33.5 MB, L2-resident
  bf16_t* WgA   = (bf16_t*)take((size_t)N3C_ * KA_ * 2);     // w_ih padded K=608
  bf16_t* WgB   = (bf16_t*)take((size_t)N3C_ * C_ * 2);      // w_hh
  bf16_t* w1bf  = (bf16_t*)take((size_t)C_ * C_ * 2);
  bf16_t* xbf   = (bf16_t*)take((size_t)B_ * LDX_ * 2);      // [frame|prev|pad|h] rows
  float*  G     = (float*) take((size_t)2 * N3C_ * B_ * 4);  // gi;gh as [n][b]
  float*  Ht    = (float*) take((size_t)C_ * B_ * 4);        // h f32 as [c][b]
  float*  Yt    = (float*) take((size_t)C_ * B_ * 4);        // tanh(h@w1+b1) as [n][b]
  unsigned* bar = (unsigned*)take(256);                      // barrier cnt + gen

  const int nEnc = B_ * S_ * I_;
  k_cast_bf16<<<(nEnc + 255) / 256, 256, 0, stream>>>(enc, encbf, nEnc);
  k_cast_bf16<<<(N3C_ * C_ + 255) / 256, 256, 0, stream>>>(w_hh, WgB, N3C_ * C_);
  k_cast_bf16<<<(C_ * C_ + 255) / 256, 256, 0, stream>>>(w1, w1bf, C_ * C_);
  k_prep_wih<<<(N3C_ * KA_ + 255) / 256, 256, 0, stream>>>(w_ih, WgA);
  k_init<<<(B_ * LDX_ + 255) / 256, 256, 0, stream>>>(xbf, Ht, bar);

  aligner_persistent<<<NWG, NTHR, 0, stream>>>(gt, mask, b_ih, b_hh, b1, w2, b2,
      encbf, WgA, WgB, w1bf, xbf, G, Ht, Yt, bar, out);
}
